// GCN1_64372969832703
// MI455X (gfx1250) — compile-verified, matrix operand order checked by reading, not run on hardware
//
#include <hip/hip_runtime.h>
#include <math.h>

typedef __attribute__((ext_vector_type(16))) __bf16 v16bf;
typedef __attribute__((ext_vector_type(8)))  float  v8f;

#define N_NODES 100000
#define N_EDGES 1600000
#define IN_CH   256
#define OUT_CH  64

// round-to-nearest-even f32 -> bf16, packed pair into one dword
__device__ __forceinline__ unsigned int pack_bf16x2(float a, float b) {
    unsigned int ua = __float_as_uint(a);
    unsigned int ub = __float_as_uint(b);
    ua += 0x7FFFu + ((ua >> 16) & 1u);
    ub += 0x7FFFu + ((ub >> 16) & 1u);
    return (ua >> 16) | (ub & 0xFFFF0000u);
}

// ---------------------------------------------------------------------------
// 1) degree init: deg[i] = 1 (self loop)
__global__ __launch_bounds__(256) void k_deg_init(float* __restrict__ deg) {
    int i = blockIdx.x * 256 + threadIdx.x;
    if (i < N_NODES) deg[i] = 1.0f;
}

// 2) deg[dst] += 1 per edge
__global__ __launch_bounds__(256) void k_edge_deg(const long long* __restrict__ dst,
                                                  float* __restrict__ deg) {
    int e = blockIdx.x * 256 + threadIdx.x;
    if (e < N_EDGES) atomicAdd(&deg[(int)dst[e]], 1.0f);
}

// 3) h = x @ W via v_wmma_f32_16x16x32_bf16.
//    256 threads = 8 waves; each wave owns a 16x64 output tile (128 rows/block).
__global__ __launch_bounds__(256) void k_gemm_xw(const float* __restrict__ x,
                                                 const float* __restrict__ W,
                                                 float* __restrict__ h) {
    // W transposed + bf16-pair packed: pair p = [n][k/2] holds W[k][n], W[k+1][n]
    __shared__ unsigned int sW[OUT_CH * (IN_CH / 2)];          // 32 KB
    __shared__ unsigned int sX[8][16 * (IN_CH / 2)];           // 64 KB (per-wave x tile)

    const int tid  = threadIdx.x;
    const int wave = tid >> 5;
    const int lane = tid & 31;
    const int half = lane >> 4;
    const int l16  = lane & 15;

    // stage W -> LDS (bf16, column-packed so each B fragment dword is contiguous)
    for (int p = tid; p < OUT_CH * (IN_CH / 2); p += 256) {
        int n  = p >> 7;          // column
        int k  = (p & 127) << 1;  // even K
        sW[p] = pack_bf16x2(W[k * OUT_CH + n], W[(k + 1) * OUT_CH + n]);
    }

    // stage this wave's 16x256 x tile -> LDS (coalesced float4 loads, bf16 pack)
    const int rowBase = blockIdx.x * 128 + wave * 16;
    for (int i = 0; i < 32; ++i) {
        int flat = i * 128 + lane * 4;       // 0..4095
        int m = flat >> 8;                   // row within tile
        int c = flat & 255;                  // column (multiple of 4)
        int row = rowBase + m;
        float f0 = 0.f, f1 = 0.f, f2 = 0.f, f3 = 0.f;
        if (row < N_NODES) {
            const float4 v = *(const float4*)(x + (size_t)row * IN_CH + c);
            f0 = v.x; f1 = v.y; f2 = v.z; f3 = v.w;
        }
        unsigned int* dp = &sX[wave][m * (IN_CH / 2) + (c >> 1)];
        dp[0] = pack_bf16x2(f0, f1);
        dp[1] = pack_bf16x2(f2, f3);
    }
    __syncthreads();

    v8f acc[4] = {};
    for (int kt = 0; kt < IN_CH / 32; ++kt) {
        const int kb = kt * 32 + 8 * half;

        union { unsigned int u[8]; v16bf v; } A;
        #pragma unroll
        for (int vv = 0; vv < 8; ++vv) {
            int kk = ((vv < 4) ? (2 * vv) : (2 * vv + 8)) + kb;   // ISA A layout
            A.u[vv] = sX[wave][l16 * (IN_CH / 2) + (kk >> 1)];
        }

        #pragma unroll
        for (int nt = 0; nt < 4; ++nt) {
            int n = nt * 16 + l16;
            union { unsigned int u[8]; v16bf v; } B;
            #pragma unroll
            for (int vv = 0; vv < 8; ++vv) {
                int kk = ((vv < 4) ? (2 * vv) : (2 * vv + 8)) + kb; // ISA B layout
                B.u[vv] = sW[n * (IN_CH / 2) + (kk >> 1)];
            }
            acc[nt] = __builtin_amdgcn_wmma_f32_16x16x32_bf16(
                false, A.v, false, B.v, (short)0, acc[nt], false, false);
        }
    }

    // C/D layout: VGPR r -> row (r + 8*half), col = nt*16 + l16
    #pragma unroll
    for (int nt = 0; nt < 4; ++nt) {
        int n = nt * 16 + l16;
        #pragma unroll
        for (int r = 0; r < 8; ++r) {
            int row = rowBase + r + 8 * half;
            if (row < N_NODES) h[(size_t)row * OUT_CH + n] = acc[nt][r];
        }
    }
}

// 4) dinv = rsqrt(deg) in place (deg >= 1 always)
__global__ __launch_bounds__(256) void k_dinv(float* __restrict__ deg) {
    int i = blockIdx.x * 256 + threadIdx.x;
    if (i < N_NODES) deg[i] = rsqrtf(deg[i]);
}

// 5) out = h * dinv^2 (self loop) + bias  (fully overwrites d_out)
__global__ __launch_bounds__(256) void k_agg_init(const float* __restrict__ h,
                                                  const float* __restrict__ dinv,
                                                  const float* __restrict__ b,
                                                  float* __restrict__ out) {
    long long t = (long long)blockIdx.x * 256 + threadIdx.x;
    if (t >= (long long)N_NODES * OUT_CH) return;
    int row = (int)(t >> 6);
    int c   = (int)(t & 63);
    float dv = dinv[row];
    out[t] = h[t] * dv * dv + b[c];
}

// 6) scatter-add messages: one wave per edge, 2 channels per lane
__global__ __launch_bounds__(256) void k_scatter(const long long* __restrict__ src,
                                                 const long long* __restrict__ dst,
                                                 const float* __restrict__ h,
                                                 const float* __restrict__ dinv,
                                                 float* __restrict__ out) {
    long long t = (long long)blockIdx.x * 256 + threadIdx.x;
    if (t >= (long long)N_EDGES * 32) return;
    int e = (int)(t >> 5);
    int c = (int)(t & 31) * 2;
    int s = (int)src[e];
    int d = (int)dst[e];
    float nrm = dinv[s] * dinv[d];
    float m0 = h[(size_t)s * OUT_CH + c]     * nrm;
    float m1 = h[(size_t)s * OUT_CH + c + 1] * nrm;
    atomicAdd(out + (size_t)d * OUT_CH + c,     m0);
    atomicAdd(out + (size_t)d * OUT_CH + c + 1, m1);
}

// 7) row-wise log_softmax over 64 channels: one wave per row, 2 ch per lane
__global__ __launch_bounds__(256) void k_log_softmax(float* __restrict__ out) {
    int wave = threadIdx.x >> 5, lane = threadIdx.x & 31;
    int row = blockIdx.x * 8 + wave;
    if (row >= N_NODES) return;
    float* p = out + (size_t)row * OUT_CH;
    float v0 = p[lane * 2], v1 = p[lane * 2 + 1];
    float m = fmaxf(v0, v1);
    #pragma unroll
    for (int o = 16; o > 0; o >>= 1) m = fmaxf(m, __shfl_xor(m, o, 32));
    float s = __expf(v0 - m) + __expf(v1 - m);
    #pragma unroll
    for (int o = 16; o > 0; o >>= 1) s += __shfl_xor(s, o, 32);
    float lse = m + __logf(s);
    p[lane * 2]     = v0 - lse;
    p[lane * 2 + 1] = v1 - lse;
}

// ---------------------------------------------------------------------------
extern "C" void kernel_launch(void* const* d_in, const int* in_sizes, int n_in,
                              void* d_out, int out_size, void* d_ws, size_t ws_size,
                              hipStream_t stream) {
    const float*     x  = (const float*)d_in[0];
    const long long* ei = (const long long*)d_in[1];   // int64 [2, E]
    const float*     W  = (const float*)d_in[2];
    const float*     b  = (const float*)d_in[3];
    float*           out = (float*)d_out;

    const long long* src = ei;
    const long long* dst = ei + N_EDGES;

    float* h   = (float*)d_ws;                          // N * 64 floats
    float* deg = h + (size_t)N_NODES * OUT_CH;          // N floats (deg -> dinv in place)

    k_deg_init<<<(N_NODES + 255) / 256, 256, 0, stream>>>(deg);
    k_edge_deg<<<(N_EDGES + 255) / 256, 256, 0, stream>>>(dst, deg);
    k_gemm_xw <<<(N_NODES + 127) / 128, 256, 0, stream>>>(x, W, h);
    k_dinv    <<<(N_NODES + 255) / 256, 256, 0, stream>>>(deg);
    {
        long long total = (long long)N_NODES * OUT_CH;
        k_agg_init<<<(int)((total + 255) / 256), 256, 0, stream>>>(h, deg, b, out);
    }
    {
        long long total = (long long)N_EDGES * 32;
        k_scatter<<<(int)((total + 255) / 256), 256, 0, stream>>>(src, dst, h, deg, out);
    }
    k_log_softmax<<<(N_NODES + 7) / 8, 256, 0, stream>>>(out);
}
